// DynamicsPredictor_70257075028096
// MI455X (gfx1250) — matrix operand order repaired and checked by reading, not compiled, not run on hardware
//
#include <hip/hip_runtime.h>
#include <math.h>

// ---------------- problem constants ----------------
#define B_      2
#define N_      2048
#define NP_     1800
#define NREL_   8192
#define SD_     3
#define NINST_  4
#define NCLUS_  50
#define NF_     200
#define MEMD_   400

// padded (multiple-of-32) leading dims for f16 GEMM operands
#define XN_LD    448   // node feature 428 -> 448
#define XE_LD    864   // edge feature 857 -> 864
#define H_LD     224   // hidden 200 -> 224 (14 N-tiles = 2 groups of 7)
#define EREL_LD  608   // [re,er,es] 600 -> 608
#define EFFIN_LD 416   // [pe,agg] 400 -> 416
#define NRM_LD   32    // final 3 -> 32 (2 N-tiles = 1 group of 2)

typedef __attribute__((ext_vector_type(16))) _Float16 v16h;
typedef __attribute__((ext_vector_type(8)))  float    v8f;

// ============================================================
// WMMA GEMM:  C = act(A[M x Kpad] * W + bias)
// A row-major f16 (lda = Kpad), W pre-packed into B-fragment layout.
// Each wave computes a 16 x (16*NT) strip: the A fragment is loaded
// once per K-chunk and reused for NT WMMAs (N register blocking),
// cutting A-side HBM traffic by NT.
// A frag (lane l, half h): M=l&15, K = 16*(l>>4) + h
// B frag (lane l, half h): N=l&15, K = 16*(l>>4) + h
// D (lane l, vgpr v):      N=l&15, M = v + 8*(l>>4)
// ============================================================
template <int NT>
__global__ __launch_bounds__(128) void k_gemm_wmma(
    const _Float16* __restrict__ A, int lda,
    const _Float16* __restrict__ Bp,
    const float* __restrict__ bias, int Nout,
    _Float16* __restrict__ C16, float* __restrict__ C32, int ldc,
    int Mtiles, int NtGroups, int Ntiles, int chunks, int relu)
{
  int wave = blockIdx.x * (blockDim.x >> 5) + (threadIdx.x >> 5);
  int lane = threadIdx.x & 31;
  if (wave >= Mtiles * NtGroups) return;
  int mt = wave / NtGroups, ng = wave % NtGroups;
  int nt0 = ng * NT;
  int kg   = lane >> 4;               // 0 or 1 (lane half-group)
  int rowA = mt * 16 + (lane & 15);

  v8f zero = {0.f,0.f,0.f,0.f,0.f,0.f,0.f,0.f};
  v8f acc[NT];
#pragma unroll
  for (int t = 0; t < NT; ++t) acc[t] = zero;

  const _Float16* aptr = A + (size_t)rowA * lda + kg * 16;
  const _Float16* bptr = Bp + ((size_t)nt0 * 32 + lane) * 16;
  size_t bstep = (size_t)Ntiles * 512;   // halves per K-chunk across all N-tiles

  for (int c = 0; c < chunks; ++c) {
    union { v16h v; uint4 q[2]; } a;
    a.q[0] = *(const uint4*)(aptr);        // K = c*32 + 16*kg + 0..7
    a.q[1] = *(const uint4*)(aptr + 8);    // K = c*32 + 16*kg + 8..15
    aptr += 32;
#pragma unroll
    for (int t = 0; t < NT; ++t) {
      v16h b = *(const v16h*)(bptr + (size_t)t * 512);
      acc[t] = __builtin_amdgcn_wmma_f32_16x16x32_f16(
          false, a.v, false, b, (short)0, acc[t], false, false);
    }
    bptr += bstep;
  }

  int rbase = mt * 16 + 8 * kg;
#pragma unroll
  for (int t = 0; t < NT; ++t) {
    int col = (nt0 + t) * 16 + (lane & 15);
    float bv = (col < Nout) ? bias[col] : 0.f;
#pragma unroll
    for (int v = 0; v < 8; ++v) {
      float val = acc[t][v] + bv;
      if (relu) val = fmaxf(val, 0.f);
      size_t o = (size_t)(rbase + v) * ldc + col;
      if (C16) C16[o] = (_Float16)val;
      if (C32) C32[o] = val;
    }
  }
}

// Pack fp32 weight W[K x Nout] into B-fragment layout, zero padded to Kpad x Npad.
__global__ void k_packw(const float* __restrict__ W, _Float16* __restrict__ P,
                        int K, int Nout, int Kpad, int Npad)
{
  int i = blockIdx.x * blockDim.x + threadIdx.x;
  int total = Kpad * Npad;
  if (i >= total) return;
  int Ntiles = Npad >> 4;
  int h    = i & 15;
  int lane = (i >> 4) & 31;
  int t    = i >> 9;             // (chunk*Ntiles + ntile)
  int ntile = t % Ntiles;
  int chunk = t / Ntiles;
  int k = chunk * 32 + 16 * (lane >> 4) + h;
  int n = ntile * 16 + (lane & 15);
  float v = (k < K && n < Nout) ? W[(size_t)k * Nout + n] : 0.f;
  P[i] = (_Float16)v;
}

// ---------------- index extraction (one-hot -> int) ----------------
__global__ void k_argmax(const float* __restrict__ R, int* __restrict__ idx, int cols)
{
  int row = blockIdx.x;
  const float* r = R + (size_t)row * cols;
  if (threadIdx.x == 0) idx[row] = 0;
  __syncthreads();
  for (int c = threadIdx.x; c < cols; c += blockDim.x)
    if (r[c] > 0.5f) idx[row] = c;
}

__global__ void k_sender_flags(const int* __restrict__ idxRs, int* __restrict__ flags)
{
  int k = blockIdx.x * blockDim.x + threadIdx.x;
  if (k >= B_ * NREL_) return;
  int b = k / NREL_;
  int s = idxRs[k];
  if (s >= 310 && s < 321) atomicOr(&flags[2 * b],     1);
  if (s >= 321)            atomicOr(&flags[2 * b + 1], 1);
}

// ---------------- state normalization: sn_t (B,N,12) ----------------
__global__ void k_snt(const float* __restrict__ state,
                      const float* __restrict__ mean_p, const float* __restrict__ std_p,
                      const float* __restrict__ mean_d, const float* __restrict__ std_d,
                      float* __restrict__ snt)
{
  int i = blockIdx.x * blockDim.x + threadIdx.x;
  if (i >= B_ * N_ * 12) return;
  int d = i % 3, h = (i / 3) % 4, n = (i / 12) % N_, b = i / (12 * N_);
  float v;
  if (h < 3) {
    float s1 = state[(((size_t)b * 4 + h + 1) * N_ + n) * 3 + d];
    float s0 = state[(((size_t)b * 4 + h)     * N_ + n) * 3 + d];
    v = (s1 - s0 - mean_d[d]) / std_d[d];
    if (n < 301) v = 0.f;
  } else {
    v = (state[(((size_t)b * 4 + 3) * N_ + n) * 3 + d] - mean_p[d]) / std_p[d];
  }
  snt[i] = v;
}

// ---------------- per-instance center (B,4,12) ----------------
__global__ void k_instcenter(const float* __restrict__ snt, const int* __restrict__ instId,
                             float* __restrict__ instC)
{
  int b = blockIdx.x / NINST_, inst = blockIdx.x % NINST_;
  int t = threadIdx.x;
  __shared__ float cnt;
  if (t == 12) {
    int c = 0;
    for (int p = 0; p < NP_; ++p) if (instId[b * NP_ + p] == inst) ++c;
    cnt = (float)c;
  }
  __syncthreads();
  if (t < 12) {
    float s = 0.f;
    for (int p = 0; p < NP_; ++p)
      if (instId[b * NP_ + p] == inst) s += snt[((size_t)b * N_ + p) * 12 + t];
    instC[(b * NINST_ + inst) * 12 + t] = s / (cnt + 1e-6f);
  }
}

// ---------------- node features Xn (B*N, 448) f16 ----------------
__global__ void k_xn(const float* __restrict__ attrs, const float* __restrict__ physics,
                     const float* __restrict__ memory, const float* __restrict__ snt,
                     const float* __restrict__ instC, const int* __restrict__ instId,
                     const float* __restrict__ p_rigid, _Float16* __restrict__ Xn)
{
  int node = blockIdx.x;
  int b = node / N_, n = node % N_;
  float prpp = 0.f; int iid = 0; bool isP = (n < NP_);
  if (isP) { iid = instId[b * NP_ + n]; prpp = p_rigid[b * NINST_ + iid]; }
  const float* sn = snt + (size_t)node * 12;
  _Float16* out = Xn + (size_t)node * XN_LD;
  for (int col = threadIdx.x; col < XN_LD; col += blockDim.x) {
    float v;
    if (col < 3)        v = attrs[(size_t)node * 3 + col];
    else if (col == 3)  v = isP ? physics[b * NP_ + n] : 0.f;
    else if (col < 16) {
      if (isP) {
        int d = col - 4;
        float sv  = sn[d];
        float cpp = instC[(b * NINST_ + iid) * 12 + d];
        v = sv - ((1.f - prpp) * sv + prpp * cpp);   // offset = sn - c
      } else v = 0.f;
    }
    else if (col < 416) v = memory[((size_t)b * MEMD_ + (col - 16)) * N_ + n];
    else if (col < 428) v = sn[col - 416];
    else                v = 0.f;
    out[col] = (_Float16)v;
  }
}

// ---------------- edge features Xe (B*NREL, 864) f16 ----------------
__global__ void k_xe(const _Float16* __restrict__ Xn, const int* __restrict__ idxRr,
                     const int* __restrict__ idxRs, const int* __restrict__ instId,
                     _Float16* __restrict__ Xe)
{
  int e = blockIdx.x;
  int b = e / NREL_;
  int r = idxRr[e], s = idxRs[e];
  const _Float16* xr = Xn + ((size_t)b * N_ + r) * XN_LD;
  const _Float16* xs = Xn + ((size_t)b * N_ + s) * XN_LD;
  _Float16* out = Xe + (size_t)e * XE_LD;
  for (int col = threadIdx.x; col < XE_LD; col += blockDim.x) {
    _Float16 v;
    if (col < 416)       v = xr[col];
    else if (col < 832)  v = xs[col - 416];
    else if (col < 844)  v = xr[416 + col - 832];
    else if (col < 856)  v = xs[416 + col - 844];
    else if (col == 856) {
      int ri = (r < NP_) ? instId[b * NP_ + r] : -1;
      int si = (s < NP_) ? instId[b * NP_ + s] : -1;
      float g = (ri == si) ? 0.f : ((ri >= 0 && si >= 0) ? 2.f : 1.f);
      v = (_Float16)g;
    } else v = (_Float16)0.f;
    out[col] = v;
  }
}

// ---------------- propagation gather: [re, eff[r], eff[s]] ----------------
__global__ void k_erelin(const _Float16* __restrict__ re, const _Float16* __restrict__ eff,
                         const int* __restrict__ idxRr, const int* __restrict__ idxRs,
                         _Float16* __restrict__ out)
{
  int e = blockIdx.x;
  int b = e / NREL_;
  int r = idxRr[e], s = idxRs[e];
  const _Float16* rr = re  + (size_t)e * H_LD;
  const _Float16* er = eff + ((size_t)b * N_ + r) * H_LD;
  const _Float16* es = eff + ((size_t)b * N_ + s) * H_LD;
  _Float16* o = out + (size_t)e * EREL_LD;
  for (int col = threadIdx.x; col < EREL_LD; col += blockDim.x) {
    _Float16 v;
    if (col < 200)      v = rr[col];
    else if (col < 400) v = er[col - 200];
    else if (col < 600) v = es[col - 400];
    else                v = (_Float16)0.f;
    o[col] = v;
  }
}

__global__ void k_scatter(const float* __restrict__ erel, const int* __restrict__ idxRr,
                          float* __restrict__ agg)
{
  int e = blockIdx.x, f = threadIdx.x;
  if (f >= NF_) return;
  int b = e / NREL_;
  int r = idxRr[e];
  atomicAdd(&agg[((size_t)b * N_ + r) * H_LD + f], erel[(size_t)e * H_LD + f]);
}

__global__ void k_effin(const _Float16* __restrict__ pe, const float* __restrict__ agg,
                        _Float16* __restrict__ out)
{
  int node = blockIdx.x;
  _Float16* o = out + (size_t)node * EFFIN_LD;
  for (int col = threadIdx.x; col < EFFIN_LD; col += blockDim.x) {
    _Float16 v;
    if (col < 200)      v = pe[(size_t)node * H_LD + col];
    else if (col < 400) v = (_Float16)agg[(size_t)node * H_LD + (col - 200)];
    else                v = (_Float16)0.f;
    o[col] = v;
  }
}

// ---------------- particle-row gather, instance reduction ----------------
__global__ void k_effp(const _Float16* __restrict__ eff, _Float16* __restrict__ effP)
{
  int pr = blockIdx.x;
  int b = pr / NP_, p = pr % NP_;
  const _Float16* src = eff  + ((size_t)b * N_ + p) * H_LD;
  _Float16* dst       = effP + (size_t)pr * H_LD;
  for (int c = threadIdx.x; c < H_LD; c += blockDim.x) dst[c] = src[c];
}

__global__ void k_insteff(const _Float16* __restrict__ effP, const int* __restrict__ instId,
                          float* __restrict__ instEff)
{
  int b = blockIdx.x / NINST_, inst = blockIdx.x % NINST_;
  int f = threadIdx.x;
  if (f >= NF_) return;
  float s = 0.f;
  for (int p = 0; p < NP_; ++p)
    if (instId[b * NP_ + p] == inst) s += (float)effP[((size_t)b * NP_ + p) * H_LD + f];
  instEff[(b * NINST_ + inst) * NF_ + f] = s;
}

// ---------------- tiny rigid MLP (8 rows): 200->200->200->7 ----------------
__global__ void k_rgmlp(const float* __restrict__ instEff,
                        const float* __restrict__ w0, const float* __restrict__ b0,
                        const float* __restrict__ w1, const float* __restrict__ b1,
                        const float* __restrict__ w2, const float* __restrict__ b2,
                        float* __restrict__ rig)
{
  __shared__ float h0[NF_], h1[NF_];
  int r = blockIdx.x;
  const float* x = instEff + r * NF_;
  for (int j = threadIdx.x; j < NF_; j += blockDim.x) {
    float a = b0[j];
    for (int k = 0; k < NF_; ++k) a += x[k] * w0[k * NF_ + j];
    h0[j] = fmaxf(a, 0.f);
  }
  __syncthreads();
  for (int j = threadIdx.x; j < NF_; j += blockDim.x) {
    float a = b1[j];
    for (int k = 0; k < NF_; ++k) a += h0[k] * w1[k * NF_ + j];
    h1[j] = fmaxf(a, 0.f);
  }
  __syncthreads();
  for (int j = threadIdx.x; j < 7; j += blockDim.x) {
    float a = b2[j];
    for (int k = 0; k < NF_; ++k) a += h1[k] * w2[k * 7 + j];
    rig[r * 7 + j] = a;
  }
}

// ---------------- quaternion -> R, cc, bvec (per instance) ----------------
__global__ void k_rigidparams(const float* __restrict__ rig, const float* __restrict__ instC,
                              const float* __restrict__ mean_p, const float* __restrict__ std_p,
                              const float* __restrict__ mean_d, const float* __restrict__ std_d,
                              float* __restrict__ rp)
{
  int r = blockIdx.x * blockDim.x + threadIdx.x;
  if (r >= B_ * NINST_) return;
  float q0 = rig[r * 7 + 0] + 1.f, q1 = rig[r * 7 + 1];
  float q2 = rig[r * 7 + 2],       q3 = rig[r * 7 + 3];
  float inv = 1.f / sqrtf(q0 * q0 + q1 * q1 + q2 * q2 + q3 * q3);
  float w = q0 * inv, x = q1 * inv, y = q2 * inv, z = q3 * inv;
  float* o = rp + r * 15;
  o[0] = 1.f - 2.f*y*y - 2.f*z*z;  o[1] = 2.f*x*y + 2.f*z*w;  o[2] = 2.f*x*z - 2.f*y*w;
  o[3] = 2.f*x*y - 2.f*z*w;  o[4] = 1.f - 2.f*x*x - 2.f*z*z;  o[5] = 2.f*y*z + 2.f*x*w;
  o[6] = 2.f*x*z + 2.f*y*w;  o[7] = 2.f*y*z - 2.f*x*w;  o[8] = 1.f - 2.f*x*x - 2.f*y*y;
  for (int d = 0; d < 3; ++d) {
    o[9 + d]  = instC[r * 12 + 9 + d] * std_p[d] + mean_p[d];   // cc (center, "cur" block)
    o[12 + d] = rig[r * 7 + 4 + d]    * std_d[d] + mean_d[d];   // bvec
  }
}

// ---------------- final prediction select -> d_out ----------------
__global__ void k_pred(const float* __restrict__ state, const int* __restrict__ instId,
                       const float* __restrict__ rp, const int* __restrict__ flags,
                       const float* __restrict__ nrm,
                       const float* __restrict__ mean_d, const float* __restrict__ std_d,
                       float* __restrict__ out)
{
  int i = blockIdx.x * blockDim.x + threadIdx.x;
  if (i >= B_ * NP_) return;
  int b = i / NP_, p = i % NP_;
  int ngt = (flags[2 * b] ? 1 : 0) + (flags[2 * b + 1] ? 1 : 0);
  float p0[3];
  for (int d = 0; d < 3; ++d) p0[d] = state[(((size_t)b * 4 + 3) * N_ + p) * 3 + d];
  const float* o = rp + (b * NINST_ + instId[i]) * 15;
  float t[3];
  for (int d = 0; d < 3; ++d) t[d] = p0[d] - o[9 + d];
  for (int c = 0; c < 3; ++c) {
    float p1 = t[0] * o[c] + t[1] * o[3 + c] + t[2] * o[6 + c] + o[12 + c] + o[9 + c];
    float rigid = (p1 - p0[c] - mean_d[c]) / std_d[c];
    float station = (0.f - mean_d[c]) / std_d[c];
    float v = (ngt == 1) ? rigid : ((ngt == 0) ? station : nrm[(size_t)i * NRM_LD + c]);
    out[(size_t)i * 3 + c] = v;
  }
}

// ---------------- cluster mean/std ----------------
__global__ void k_stat_acc(const float* __restrict__ nrm, const int* __restrict__ clusId,
                           float* __restrict__ X, float* __restrict__ X2, float* __restrict__ ncnt)
{
  int i = blockIdx.x * blockDim.x + threadIdx.x;
  if (i >= B_ * NP_) return;
  int b = i / NP_;
  int c = clusId[i];
  atomicAdd(&ncnt[b * NCLUS_ + c], 1.f);
  for (int d = 0; d < 3; ++d) {
    float v = nrm[(size_t)i * NRM_LD + d];
    atomicAdd(&X [(b * 3 + d) * NCLUS_ + c], v);
    atomicAdd(&X2[(b * 3 + d) * NCLUS_ + c], v * v);
  }
}

__global__ void k_stat_final(const float* __restrict__ X, const float* __restrict__ X2,
                             const float* __restrict__ ncnt, float* __restrict__ out)
{
  __shared__ float sm[256];
  float s = 0.f;
  for (int i = threadIdx.x; i < B_ * 3 * NCLUS_; i += blockDim.x) {
    int b = i / (3 * NCLUS_);
    int c = i % NCLUS_;
    float n = ncnt[b * NCLUS_ + c];
    float m = X[i] / n;
    s += sqrtf(X2[i] / n - m * m + 1e-6f);
  }
  sm[threadIdx.x] = s;
  __syncthreads();
  for (int off = 128; off > 0; off >>= 1) {
    if (threadIdx.x < off) sm[threadIdx.x] += sm[threadIdx.x + off];
    __syncthreads();
  }
  if (threadIdx.x == 0) out[0] = sm[0] / (float)(B_ * 3 * NCLUS_);
}

// ============================================================
// host side
// ============================================================
static inline void gemmL(hipStream_t s, const _Float16* A, int lda, const _Float16* Bp,
                         const float* bias, int Nout, _Float16* C16, float* C32, int ldc,
                         int M, int Npad, int Kpad, int relu)
{
  int Mt = M / 16, Nt = Npad / 16, ch = Kpad / 32;
  if (Nt % 7 == 0) {
    int groups = Nt / 7;
    int tiles = Mt * groups;
    k_gemm_wmma<7><<<(tiles + 3) / 4, 128, 0, s>>>(A, lda, Bp, bias, Nout, C16, C32, ldc,
                                                   Mt, groups, Nt, ch, relu);
  } else {
    int groups = Nt / 2;
    int tiles = Mt * groups;
    k_gemm_wmma<2><<<(tiles + 3) / 4, 128, 0, s>>>(A, lda, Bp, bias, Nout, C16, C32, ldc,
                                                   Mt, groups, Nt, ch, relu);
  }
}

static inline void packL(hipStream_t s, const float* W, _Float16* P,
                         int K, int Nout, int Kpad, int Npad)
{
  int total = Kpad * Npad;
  k_packw<<<(total + 255) / 256, 256, 0, s>>>(W, P, K, Nout, Kpad, Npad);
}

extern "C" void kernel_launch(void* const* d_in, const int* in_sizes, int n_in,
                              void* d_out, int out_size, void* d_ws, size_t ws_size,
                              hipStream_t stream)
{
  (void)in_sizes; (void)n_in; (void)out_size; (void)ws_size;
  // --- inputs (setup_inputs order) ---
  const float* attrs   = (const float*)d_in[0];
  const float* state   = (const float*)d_in[1];
  const float* Rr      = (const float*)d_in[2];
  const float* Rs      = (const float*)d_in[3];
  // d_in[4] = Rn (unused by the forward pass)
  const float* memory  = (const float*)d_in[5];
  const float* p_rigid = (const float*)d_in[6];
  const float* p_inst  = (const float*)d_in[7];
  const float* physics = (const float*)d_in[8];
  const float* cluster = (const float*)d_in[9];
  const float* mean_p  = (const float*)d_in[10];
  const float* std_p   = (const float*)d_in[11];
  const float* mean_d  = (const float*)d_in[12];
  const float* std_d   = (const float*)d_in[13];
  // params flattened as sorted pytree: nr, pe, pp, re, rg, rp ; inner keys sorted
  const float* nr_b0 = (const float*)d_in[14];
  const float* nr_b1 = (const float*)d_in[15];
  const float* nr_b2 = (const float*)d_in[16];
  const float* nr_w0 = (const float*)d_in[17];
  const float* nr_w1 = (const float*)d_in[18];
  const float* nr_w2 = (const float*)d_in[19];
  const float* pe_b0 = (const float*)d_in[20];
  const float* pe_b1 = (const float*)d_in[21];
  const float* pe_b2 = (const float*)d_in[22];
  const float* pe_w0 = (const float*)d_in[23];
  const float* pe_w1 = (const float*)d_in[24];
  const float* pe_w2 = (const float*)d_in[25];
  const float* pp_b  = (const float*)d_in[26];
  const float* pp_w  = (const float*)d_in[27];
  const float* re_b0 = (const float*)d_in[28];
  const float* re_b1 = (const float*)d_in[29];
  const float* re_b2 = (const float*)d_in[30];
  const float* re_w0 = (const float*)d_in[31];
  const float* re_w1 = (const float*)d_in[32];
  const float* re_w2 = (const float*)d_in[33];
  const float* rg_b0 = (const float*)d_in[34];
  const float* rg_b1 = (const float*)d_in[35];
  const float* rg_b2 = (const float*)d_in[36];
  const float* rg_w0 = (const float*)d_in[37];
  const float* rg_w1 = (const float*)d_in[38];
  const float* rg_w2 = (const float*)d_in[39];
  const float* rp_b  = (const float*)d_in[40];
  const float* rp_w  = (const float*)d_in[41];

  float* outPred = (float*)d_out;                  // (B, N_P, 3)
  float* outStd  = outPred + (size_t)B_ * NP_ * 3; // scalar

  // --- workspace carve-out (256B aligned) ---
  char* base = (char*)d_ws;
  size_t off = 0;
  auto alloc = [&](size_t bytes) -> void* {
    off = (off + 255) & ~(size_t)255;
    void* p = base + off;
    off += bytes;
    return p;
  };
  int*      idxRr  = (int*)alloc((size_t)B_ * NREL_ * 4);
  int*      idxRs  = (int*)alloc((size_t)B_ * NREL_ * 4);
  int*      instId = (int*)alloc((size_t)B_ * NP_ * 4);
  int*      clusId = (int*)alloc((size_t)B_ * NP_ * 4);
  int*      flags  = (int*)alloc(2 * B_ * 4);
  float*    snt    = (float*)alloc((size_t)B_ * N_ * 12 * 4);
  float*    instC  = (float*)alloc((size_t)B_ * NINST_ * 12 * 4);
  _Float16* Xn     = (_Float16*)alloc((size_t)B_ * N_ * XN_LD * 2);
  _Float16* Xe     = (_Float16*)alloc((size_t)B_ * NREL_ * XE_LD * 2);
  _Float16* hbuf1  = (_Float16*)alloc((size_t)B_ * NREL_ * H_LD * 2);
  _Float16* hbuf2  = (_Float16*)alloc((size_t)B_ * NREL_ * H_LD * 2);
  _Float16* peB    = (_Float16*)alloc((size_t)B_ * N_ * H_LD * 2);
  _Float16* reB    = (_Float16*)alloc((size_t)B_ * NREL_ * H_LD * 2);
  _Float16* effB   = (_Float16*)alloc((size_t)B_ * N_ * H_LD * 2);
  float*    erelF  = (float*)alloc((size_t)B_ * NREL_ * H_LD * 4);
  float*    agg    = (float*)alloc((size_t)B_ * N_ * H_LD * 4);
  _Float16* effin  = (_Float16*)alloc((size_t)B_ * N_ * EFFIN_LD * 2);
  _Float16* effP   = (_Float16*)alloc((size_t)B_ * NP_ * H_LD * 2);
  float*    nrmB   = (float*)alloc((size_t)B_ * NP_ * NRM_LD * 4);
  float*    instEff= (float*)alloc((size_t)B_ * NINST_ * NF_ * 4);
  float*    rig    = (float*)alloc((size_t)B_ * NINST_ * 7 * 4);
  float*    rpar   = (float*)alloc((size_t)B_ * NINST_ * 15 * 4);
  float*    Xst    = (float*)alloc((size_t)B_ * 3 * NCLUS_ * 4);
  float*    X2st   = (float*)alloc((size_t)B_ * 3 * NCLUS_ * 4);
  float*    ncnt   = (float*)alloc((size_t)B_ * NCLUS_ * 4);
  // packed weights (f16, B-fragment layout)
  _Float16* peW0p = (_Float16*)alloc((size_t)XN_LD * H_LD * 2);
  _Float16* peW1p = (_Float16*)alloc((size_t)H_LD * H_LD * 2);
  _Float16* peW2p = (_Float16*)alloc((size_t)H_LD * H_LD * 2);
  _Float16* reW0p = (_Float16*)alloc((size_t)XE_LD * H_LD * 2);
  _Float16* reW1p = (_Float16*)alloc((size_t)H_LD * H_LD * 2);
  _Float16* reW2p = (_Float16*)alloc((size_t)H_LD * H_LD * 2);
  _Float16* rpWp  = (_Float16*)alloc((size_t)EREL_LD * H_LD * 2);
  _Float16* ppWp  = (_Float16*)alloc((size_t)EFFIN_LD * H_LD * 2);
  _Float16* nrW0p = (_Float16*)alloc((size_t)H_LD * H_LD * 2);
  _Float16* nrW1p = (_Float16*)alloc((size_t)H_LD * H_LD * 2);
  _Float16* nrW2p = (_Float16*)alloc((size_t)H_LD * NRM_LD * 2);

  // --- 0) pack weights ---
  packL(stream, pe_w0, peW0p, 428, NF_, XN_LD, H_LD);
  packL(stream, pe_w1, peW1p, NF_, NF_, H_LD, H_LD);
  packL(stream, pe_w2, peW2p, NF_, NF_, H_LD, H_LD);
  packL(stream, re_w0, reW0p, 857, NF_, XE_LD, H_LD);
  packL(stream, re_w1, reW1p, NF_, NF_, H_LD, H_LD);
  packL(stream, re_w2, reW2p, NF_, NF_, H_LD, H_LD);
  packL(stream, rp_w,  rpWp,  600, NF_, EREL_LD, H_LD);
  packL(stream, pp_w,  ppWp,  400, NF_, EFFIN_LD, H_LD);
  packL(stream, nr_w0, nrW0p, NF_, NF_, H_LD, H_LD);
  packL(stream, nr_w1, nrW1p, NF_, NF_, H_LD, H_LD);
  packL(stream, nr_w2, nrW2p, NF_, 3,   H_LD, NRM_LD);

  // --- 1) recover indices from dense one-hots (turn matmuls into gathers) ---
  k_argmax<<<B_ * NREL_, 256, 0, stream>>>(Rr, idxRr, N_);
  k_argmax<<<B_ * NREL_, 256, 0, stream>>>(Rs, idxRs, N_);
  k_argmax<<<B_ * NP_,    64, 0, stream>>>(p_inst, instId, NINST_);
  k_argmax<<<B_ * NP_,    64, 0, stream>>>(cluster, clusId, NCLUS_);
  hipMemsetAsync(flags, 0, 2 * B_ * 4, stream);
  k_sender_flags<<<(B_ * NREL_ + 255) / 256, 256, 0, stream>>>(idxRs, flags);

  // --- 2) feature construction ---
  k_snt<<<(B_ * N_ * 12 + 255) / 256, 256, 0, stream>>>(state, mean_p, std_p, mean_d, std_d, snt);
  k_instcenter<<<B_ * NINST_, 32, 0, stream>>>(snt, instId, instC);
  k_xn<<<B_ * N_, 256, 0, stream>>>(attrs, physics, memory, snt, instC, instId, p_rigid, Xn);
  k_xe<<<B_ * NREL_, 256, 0, stream>>>(Xn, idxRr, idxRs, instId, Xe);

  // --- 3) particle encoder pe: 448 -> 224 -> 224 -> 224 (relu) ---
  gemmL(stream, Xn,    XN_LD, peW0p, pe_b0, NF_, hbuf1, nullptr, H_LD, B_ * N_, H_LD, XN_LD, 1);
  gemmL(stream, hbuf1, H_LD,  peW1p, pe_b1, NF_, hbuf2, nullptr, H_LD, B_ * N_, H_LD, H_LD, 1);
  gemmL(stream, hbuf2, H_LD,  peW2p, pe_b2, NF_, peB,   nullptr, H_LD, B_ * N_, H_LD, H_LD, 1);

  // --- 4) relation encoder re: 864 -> 224 -> 224 -> 224 (relu) ---
  gemmL(stream, Xe,    XE_LD, reW0p, re_b0, NF_, hbuf1, nullptr, H_LD, B_ * NREL_, H_LD, XE_LD, 1);
  gemmL(stream, hbuf1, H_LD,  reW1p, re_b1, NF_, hbuf2, nullptr, H_LD, B_ * NREL_, H_LD, H_LD, 1);
  gemmL(stream, hbuf2, H_LD,  reW2p, re_b2, NF_, reB,   nullptr, H_LD, B_ * NREL_, H_LD, H_LD, 1);

  // --- 5) propagation (3 steps); Xe is dead now -> reuse it for erel input ---
  _Float16* erelIn = Xe;
  for (int step = 0; step < 3; ++step) {
    const _Float16* curEff = (step == 0) ? peB : effB;
    k_erelin<<<B_ * NREL_, 256, 0, stream>>>(reB, curEff, idxRr, idxRs, erelIn);
    gemmL(stream, erelIn, EREL_LD, rpWp, rp_b, NF_, nullptr, erelF, H_LD,
          B_ * NREL_, H_LD, EREL_LD, 1);
    hipMemsetAsync(agg, 0, (size_t)B_ * N_ * H_LD * 4, stream);
    k_scatter<<<B_ * NREL_, 256, 0, stream>>>(erelF, idxRr, agg);
    k_effin<<<B_ * N_, 256, 0, stream>>>(peB, agg, effin);
    gemmL(stream, effin, EFFIN_LD, ppWp, pp_b, NF_, effB, nullptr, H_LD,
          B_ * N_, H_LD, EFFIN_LD, 1);
  }

  // --- 6) per-particle motion head nrm: 224 -> 224 -> 224 -> 32 (no relu last) ---
  k_effp<<<B_ * NP_, 224, 0, stream>>>(effB, effP);
  gemmL(stream, effP,  H_LD, nrW0p, nr_b0, NF_, hbuf1, nullptr, H_LD, B_ * NP_, H_LD, H_LD, 1);
  gemmL(stream, hbuf1, H_LD, nrW1p, nr_b1, NF_, hbuf2, nullptr, H_LD, B_ * NP_, H_LD, H_LD, 1);
  gemmL(stream, hbuf2, H_LD, nrW2p, nr_b2, 3, nullptr, nrmB, NRM_LD, B_ * NP_, NRM_LD, H_LD, 0);

  // --- 7) rigid branch ---
  k_insteff<<<B_ * NINST_, 256, 0, stream>>>(effP, instId, instEff);
  k_rgmlp<<<B_ * NINST_, 256, 0, stream>>>(instEff, rg_w0, rg_b0, rg_w1, rg_b1, rg_w2, rg_b2, rig);
  k_rigidparams<<<1, 64, 0, stream>>>(rig, instC, mean_p, std_p, mean_d, std_d, rpar);

  // --- 8) prediction select -> d_out ---
  k_pred<<<(B_ * NP_ + 255) / 256, 256, 0, stream>>>(state, instId, rpar, flags, nrmB,
                                                     mean_d, std_d, outPred);

  // --- 9) cluster mean/std -> d_out tail ---
  hipMemsetAsync(Xst,  0, (size_t)B_ * 3 * NCLUS_ * 4, stream);
  hipMemsetAsync(X2st, 0, (size_t)B_ * 3 * NCLUS_ * 4, stream);
  hipMemsetAsync(ncnt, 0, (size_t)B_ * NCLUS_ * 4, stream);
  k_stat_acc<<<(B_ * NP_ + 255) / 256, 256, 0, stream>>>(nrmB, clusId, Xst, X2st, ncnt);
  k_stat_final<<<1, 256, 0, stream>>>(Xst, X2st, ncnt, outStd);
}
